// GAT_PYNQ_30159260352622
// MI455X (gfx1250) — matrix operand order repaired and checked by reading, not compiled, not run on hardware
//
#include <hip/hip_runtime.h>
#include <hip/hip_bf16.h>

#define N_NODES 50000
#define N_EDGES 800000
#define NB      64
#define HID     64
#define NEG_SLOPE 0.2f
#define BN_EPS    1e-5f

typedef __attribute__((ext_vector_type(16))) __bf16 v16bf;
typedef __attribute__((ext_vector_type(8)))  float  v8f;

__device__ __forceinline__ unsigned pack2bf16(float lo, float hi) {
  unsigned a = __float_as_uint(lo);
  unsigned b = __float_as_uint(hi);
  a = (a + 0x7FFFu + ((a >> 16) & 1u)) >> 16;   // round-to-nearest-even
  b = (b + 0x7FFFu + ((b >> 16) & 1u)) >> 16;
  return (b << 16) | (a & 0xFFFFu);
}

// ---------------------------------------------------------------------------
// GEMM: C[M,Nout] = A[M,K] @ W[K,Nout], f32 in/out, bf16 WMMA compute.
// Block = 256 threads (8 waves); tile M=128 (16 rows/wave), N chunk = 64,
// K chunk = min(K,128). LDS: A packed bf16 k-pairs (odd dword stride),
// W pre-swizzled into (k/2, n) dword pairs so fragment loads are ds_load_b32.
// ---------------------------------------------------------------------------
__global__ __launch_bounds__(256) void gemm_bf16_wmma(
    const float* __restrict__ A, const float* __restrict__ W,
    float* __restrict__ C, int M, int K, int Nout) {
  __shared__ unsigned Alds[128 * 65];  // up to KC2=64 pairs + 1 pad
  __shared__ unsigned Wlds[64 * 66];   // KC2 x (64 + 2 pad)
  const int KC  = (K < 128) ? K : 128;
  const int KC2 = KC >> 1;
  const int aStride = KC2 + 1;
  const int wStride = 66;

  const int tid   = threadIdx.x;
  const int row0  = blockIdx.x * 128;
  const int wave  = tid >> 5;
  const int lane  = tid & 31;
  const int lmod  = lane & 15;
  const int lhalf = lane >> 4;
  const int mrow  = wave << 4;

  for (int ntc = 0; ntc < Nout; ntc += 64) {
    v8f acc[4] = {};
    for (int kc = 0; kc < K; kc += KC) {
      __syncthreads();
      // stage A tile: 128 rows x KC2 packed bf16 pairs
      for (int idx = tid; idx < 128 * KC2; idx += 256) {
        int r  = idx / KC2;
        int kp = idx - r * KC2;
        int gr = row0 + r;
        float lo = 0.f, hi = 0.f;
        if (gr < M) {
          const float* ap = A + (size_t)gr * K + kc + 2 * kp;
          lo = ap[0]; hi = ap[1];
        }
        Alds[r * aStride + kp] = pack2bf16(lo, hi);
      }
      // stage W chunk: KC2 x 64, pair of consecutive-k values per dword
      for (int idx = tid; idx < KC2 * 64; idx += 256) {
        int kp = idx >> 6;
        int n  = idx & 63;
        const float* wp = W + (size_t)(kc + 2 * kp) * Nout + ntc + n;
        Wlds[kp * wStride + n] = pack2bf16(wp[0], wp[Nout]);
      }
      __syncthreads();

      for (int ksl = 0; ksl < KC; ksl += 32) {
        union { unsigned u[8]; v16bf v; } af;
#pragma unroll
        for (int v = 0; v < 8; ++v) {
          // A 16x32 bf16 layout: k = (v>=4?16:0) + half*8 + (v&3)*2
          int kloc = ksl + ((v & 4) << 2) + (lhalf << 3) + ((v & 3) << 1);
          af.u[v] = Alds[(mrow + lmod) * aStride + (kloc >> 1)];
        }
#pragma unroll
        for (int t = 0; t < 4; ++t) {
          union { unsigned u[8]; v16bf v; } bfr;
#pragma unroll
          for (int v = 0; v < 8; ++v) {
            // B 32x16 bf16 layout: k = (v>=4?16:0) + half*8 + (v&3)*2
            int kloc = ksl + ((v & 4) << 2) + (lhalf << 3) + ((v & 3) << 1);
            bfr.u[v] = Wlds[(kloc >> 1) * wStride + (t << 4) + lmod];
          }
          acc[t] = __builtin_amdgcn_wmma_f32_16x16x32_bf16(
              false, af.v, false, bfr.v, (short)0, acc[t], false, false);
        }
      }
    }
    // store C: VGPR r -> row m = r + half*8
#pragma unroll
    for (int t = 0; t < 4; ++t) {
      int n = ntc + (t << 4) + lmod;
#pragma unroll
      for (int r = 0; r < 8; ++r) {
        int gr = row0 + mrow + (lhalf << 3) + r;
        if (gr < M) C[(size_t)gr * Nout + n] = acc[t][r];
      }
    }
  }
}

// ---------------------------------------------------------------------------
// Per-(node,head) attention scores: s_src = <xW[n,h,:], a_src[h]>, same dst.
// ---------------------------------------------------------------------------
__global__ __launch_bounds__(256) void node_scores_k(
    const float* __restrict__ xW, const float* __restrict__ a_src,
    const float* __restrict__ a_dst, float* __restrict__ s_src,
    float* __restrict__ s_dst, int nh, int Hh) {
  int i = blockIdx.x * 256 + threadIdx.x;
  if (i >= nh) return;
  int h = i % Hh;                         // i = n*Hh + h
  const float* xp = xW + (size_t)i * HID; // == (n*Hh + h)*HID
  const float* as = a_src + h * HID;
  const float* ad = a_dst + h * HID;
  float ss = 0.f, sd = 0.f;
#pragma unroll 4
  for (int f = 0; f < HID; ++f) { float v = xp[f]; ss += v * as[f]; sd += v * ad[f]; }
  s_src[i] = ss; s_dst[i] = sd;
}

// order-preserving encode for float atomicMax over uint (handles negatives)
__device__ __forceinline__ unsigned ord_enc(float f) {
  unsigned u = __float_as_uint(f);
  return (u & 0x80000000u) ? ~u : (u | 0x80000000u);
}

__global__ __launch_bounds__(256) void edge_max_k(
    const int* __restrict__ src, const int* __restrict__ dst,
    const float* __restrict__ s_src, const float* __restrict__ s_dst,
    unsigned* __restrict__ m_enc, int Hh) {
  int i = blockIdx.x * 256 + threadIdx.x;
  if (i >= N_EDGES * Hh) return;
  int e = i / Hh, h = i - e * Hh;
  float v = s_src[src[e] * Hh + h] + s_dst[dst[e] * Hh + h];
  v = (v > 0.f) ? v : NEG_SLOPE * v;
  atomicMax(&m_enc[dst[e] * Hh + h], ord_enc(v));
}

__global__ __launch_bounds__(256) void fin_max_k(
    const unsigned* __restrict__ m_enc, float* __restrict__ m, int total) {
  int i = blockIdx.x * 256 + threadIdx.x;
  if (i >= total) return;
  unsigned u = m_enc[i];
  float f = (u & 0x80000000u) ? __uint_as_float(u & 0x7FFFFFFFu)
                              : __uint_as_float(~u);
  unsigned bits = __float_as_uint(f);
  bool fin = ((bits >> 23) & 0xFFu) != 0xFFu;   // empty segment decodes non-finite
  m[i] = fin ? f : 0.f;
}

__global__ __launch_bounds__(256) void edge_exp_k(
    const int* __restrict__ src, const int* __restrict__ dst,
    const float* __restrict__ s_src, const float* __restrict__ s_dst,
    const float* __restrict__ m, float* __restrict__ p,
    float* __restrict__ denom, int Hh) {
  int i = blockIdx.x * 256 + threadIdx.x;
  if (i >= N_EDGES * Hh) return;
  int e = i / Hh, h = i - e * Hh;
  int d = dst[e];
  float v = s_src[src[e] * Hh + h] + s_dst[d * Hh + h];
  v = (v > 0.f) ? v : NEG_SLOPE * v;
  float pe = __expf(v - m[d * Hh + h]);
  p[i] = pe;
  atomicAdd(&denom[d * Hh + h], pe);
}

__global__ __launch_bounds__(256) void edge_agg_k(
    const int* __restrict__ src, const int* __restrict__ dst,
    const float* __restrict__ p, const float* __restrict__ denom,
    const float* __restrict__ xW, float* __restrict__ out, int Hh) {
  int chunks = (Hh * HID) >> 2;                 // float4 chunks per edge
  long i = (long)blockIdx.x * 256 + threadIdx.x;
  if (i >= (long)N_EDGES * chunks) return;
  int e  = (int)(i / chunks);
  int c  = (int)(i - (long)e * chunks);
  int f0 = c << 2;
  int h  = f0 >> 6;                             // HID = 64
  int s = src[e], d = dst[e];
  float den = denom[d * Hh + h];
  den = den > 1e-16f ? den : 1e-16f;
  float alpha = p[(long)e * Hh + h] / den;
  const float4 xv = *reinterpret_cast<const float4*>(xW + (size_t)s * Hh * HID + f0);
  float* op = out + (size_t)d * Hh * HID + f0;
  atomicAdd(op + 0, alpha * xv.x);
  atomicAdd(op + 1, alpha * xv.y);
  atomicAdd(op + 2, alpha * xv.z);
  atomicAdd(op + 3, alpha * xv.w);
}

__global__ __launch_bounds__(256) void bias_relu_k(
    float* __restrict__ y, const float* __restrict__ b, int total, int HF) {
  int i = blockIdx.x * 256 + threadIdx.x;
  if (i >= total) return;
  float v = y[i] + b[i % HF];
  y[i] = v > 0.f ? v : 0.f;
}

__global__ __launch_bounds__(256) void bn_relu_res_k(
    const float* __restrict__ y, const float* __restrict__ b,
    const float* __restrict__ g, const float* __restrict__ be,
    const float* __restrict__ res, float* __restrict__ out, int total) {
  int i = blockIdx.x * 256 + threadIdx.x;
  if (i >= total) return;
  int f = i & (HID - 1);
  float inv = rsqrtf(1.0f + BN_EPS);
  float v = (y[i] + b[f]) * g[f] * inv + be[f];
  v = v > 0.f ? v : 0.f;
  out[i] = (res ? res[i] : 0.f) + v;
}

__global__ __launch_bounds__(256) void pool_k(
    const float* __restrict__ h, const int* __restrict__ batch,
    float* __restrict__ pooled, float* __restrict__ cnt) {
  int i = blockIdx.x * 256 + threadIdx.x;
  if (i >= N_NODES * HID) return;
  int n = i >> 6, f = i & 63;
  int bb = batch[n];
  atomicAdd(&pooled[bb * HID + f], h[i]);
  if (f == 0) atomicAdd(&cnt[bb], 1.0f);
}

__global__ __launch_bounds__(256) void mlp_k(
    const float* __restrict__ pooled, const float* __restrict__ cnt,
    const float* __restrict__ Wh1, const float* __restrict__ bh1,
    const float* __restrict__ Wh2, const float* __restrict__ bh2,
    float* __restrict__ out) {
  __shared__ float hp[NB * HID];
  __shared__ float hid[NB * 128];
  for (int i = threadIdx.x; i < NB * HID; i += 256) {
    int bb = i >> 6;
    float c = cnt[bb]; c = c > 1.f ? c : 1.f;
    hp[i] = pooled[i] / c;
  }
  __syncthreads();
  for (int i = threadIdx.x; i < NB * 128; i += 256) {
    int bb = i >> 7, j = i & 127;
    float s = bh1[j];
    for (int f = 0; f < HID; ++f) s += hp[bb * HID + f] * Wh1[f * 128 + j];
    hid[i] = s > 0.f ? s : 0.f;
  }
  __syncthreads();
  for (int i = threadIdx.x; i < NB * 10; i += 256) {
    int bb = i / 10, j = i - bb * 10;
    float s = bh2[j];
    for (int f = 0; f < 128; ++f) s += hid[bb * 128 + f] * Wh2[f * 10 + j];
    out[i] = s;
  }
}

// ---------------------------------------------------------------------------
// Host orchestration
// ---------------------------------------------------------------------------
static void run_gat(const float* X, int Kin, int Hh,
                    const float* W, const float* as, const float* ad,
                    const int* src, const int* dst,
                    float* xW, float* ssrc, float* sdst, unsigned* menc,
                    float* mbuf, float* denom, float* pbuf, float* agg,
                    hipStream_t stream) {
  const int HF = Hh * HID;
  gemm_bf16_wmma<<<(N_NODES + 127) / 128, 256, 0, stream>>>(X, W, xW, N_NODES, Kin, HF);
  const int nh = N_NODES * Hh;
  node_scores_k<<<(nh + 255) / 256, 256, 0, stream>>>(xW, as, ad, ssrc, sdst, nh, Hh);
  hipMemsetAsync(menc, 0, (size_t)nh * 4, stream);
  hipMemsetAsync(denom, 0, (size_t)nh * 4, stream);
  const int eh = N_EDGES * Hh;
  edge_max_k<<<(eh + 255) / 256, 256, 0, stream>>>(src, dst, ssrc, sdst, menc, Hh);
  fin_max_k<<<(nh + 255) / 256, 256, 0, stream>>>(menc, mbuf, nh);
  edge_exp_k<<<(eh + 255) / 256, 256, 0, stream>>>(src, dst, ssrc, sdst, mbuf, pbuf, denom, Hh);
  hipMemsetAsync(agg, 0, (size_t)N_NODES * HF * 4, stream);
  long tot = (long)N_EDGES * (HF >> 2);
  edge_agg_k<<<(int)((tot + 255) / 256), 256, 0, stream>>>(src, dst, pbuf, denom, xW, agg, Hh);
}

extern "C" void kernel_launch(void* const* d_in, const int* in_sizes, int n_in,
                              void* d_out, int out_size, void* d_ws, size_t ws_size,
                              hipStream_t stream) {
  const float* x     = (const float*)d_in[1];
  const int*   ei    = (const int*)d_in[2];
  const int*   batch = (const int*)d_in[3];
  const float* W1  = (const float*)d_in[4];
  const float* as1 = (const float*)d_in[5];
  const float* ad1 = (const float*)d_in[6];
  const float* b1  = (const float*)d_in[7];
  const float* W2  = (const float*)d_in[8];
  const float* as2 = (const float*)d_in[9];
  const float* ad2 = (const float*)d_in[10];
  const float* b2  = (const float*)d_in[11];
  const float* g2  = (const float*)d_in[12];
  const float* be2 = (const float*)d_in[13];
  const float* W3  = (const float*)d_in[14];
  const float* as3 = (const float*)d_in[15];
  const float* ad3 = (const float*)d_in[16];
  const float* b3  = (const float*)d_in[17];
  const float* g3  = (const float*)d_in[18];
  const float* be3 = (const float*)d_in[19];
  const float* W4  = (const float*)d_in[20];
  const float* as4 = (const float*)d_in[21];
  const float* ad4 = (const float*)d_in[22];
  const float* b4  = (const float*)d_in[23];
  const float* g4  = (const float*)d_in[24];
  const float* be4 = (const float*)d_in[25];
  const float* Wh1 = (const float*)d_in[26];
  const float* bh1 = (const float*)d_in[27];
  const float* Wh2 = (const float*)d_in[28];
  const float* bh2 = (const float*)d_in[29];
  (void)in_sizes; (void)n_in; (void)out_size; (void)ws_size;

  const int* src = ei;
  const int* dst = ei + N_EDGES;

  // workspace layout (floats)
  float* ws = (float*)d_ws;
  size_t off = 0;
  auto alloc = [&](size_t n) { float* p = ws + off; off += n; return p; };
  float*    xW     = alloc((size_t)N_NODES * 256);  // GEMM out (reused per layer)
  float*    aggbig = alloc((size_t)N_NODES * 256);  // layer1 aggregate -> x1
  float*    hbuf   = alloc((size_t)N_NODES * HID);  // running hidden state
  float*    tmp    = alloc((size_t)N_NODES * HID);  // per-layer aggregate
  float*    ssrc   = alloc((size_t)N_NODES * 4);
  float*    sdst   = alloc((size_t)N_NODES * 4);
  unsigned* menc   = (unsigned*)alloc((size_t)N_NODES * 4);
  float*    mbuf   = alloc((size_t)N_NODES * 4);
  float*    denom  = alloc((size_t)N_NODES * 4);
  float*    pbuf   = alloc((size_t)N_EDGES * 4);
  float*    pooled = alloc((size_t)NB * HID);
  float*    cnt    = alloc((size_t)NB);

  // layer 1: GATConv(128 -> 64, heads=4, concat) + ReLU
  run_gat(x, 128, 4, W1, as1, ad1, src, dst, xW, ssrc, sdst, menc, mbuf, denom,
          pbuf, aggbig, stream);
  bias_relu_k<<<((N_NODES * 256) + 255) / 256, 256, 0, stream>>>(
      aggbig, b1, N_NODES * 256, 256);                       // aggbig == x1

  // layer 2: GATConv(256 -> 64) + BN + ReLU (no residual)
  run_gat(aggbig, 256, 1, W2, as2, ad2, src, dst, xW, ssrc, sdst, menc, mbuf,
          denom, pbuf, tmp, stream);
  bn_relu_res_k<<<((N_NODES * HID) + 255) / 256, 256, 0, stream>>>(
      tmp, b2, g2, be2, nullptr, hbuf, N_NODES * HID);

  // residual block 1
  run_gat(hbuf, HID, 1, W3, as3, ad3, src, dst, xW, ssrc, sdst, menc, mbuf,
          denom, pbuf, tmp, stream);
  bn_relu_res_k<<<((N_NODES * HID) + 255) / 256, 256, 0, stream>>>(
      tmp, b3, g3, be3, hbuf, hbuf, N_NODES * HID);

  // residual block 2
  run_gat(hbuf, HID, 1, W4, as4, ad4, src, dst, xW, ssrc, sdst, menc, mbuf,
          denom, pbuf, tmp, stream);
  bn_relu_res_k<<<((N_NODES * HID) + 255) / 256, 256, 0, stream>>>(
      tmp, b4, g4, be4, hbuf, hbuf, N_NODES * HID);

  // global mean pool + MLP head
  hipMemsetAsync(pooled, 0, (size_t)(NB * HID + NB) * 4, stream);  // pooled + cnt
  pool_k<<<((N_NODES * HID) + 255) / 256, 256, 0, stream>>>(hbuf, batch, pooled, cnt);
  mlp_k<<<1, 256, 0, stream>>>(pooled, cnt, Wh1, bh1, Wh2, bh2, (float*)d_out);
}
